// GRU_72559177498741
// MI455X (gfx1250) — matrix operand order repaired and checked by reading, not compiled, not run on hardware
//
#include <hip/hip_runtime.h>

typedef __attribute__((ext_vector_type(16))) __bf16 v16bf;
typedef __attribute__((ext_vector_type(8)))  __bf16 v8bf;
typedef __attribute__((ext_vector_type(4)))  __bf16 v4bf;
typedef __attribute__((ext_vector_type(8)))  float  v8f;
typedef __attribute__((ext_vector_type(4)))  float  f32x4;

#define B_   64
#define T_   512
#define D_   1024
#define H_   1024
#define K3H  3072   // 3*H

#define V8F_ZERO ((v8f){0.f,0.f,0.f,0.f,0.f,0.f,0.f,0.f})

// ---------- helpers ----------

static __device__ __forceinline__ __bf16 cvt_bf16(float f) {
  unsigned u = __builtin_bit_cast(unsigned, f);
  u += 0x7FFFu + ((u >> 16) & 1u);            // round-to-nearest-even
  unsigned short s = (unsigned short)(u >> 16);
  return __builtin_bit_cast(__bf16, s);
}

// A-fragment (16x32 bf16, contiguous K order k = 16h + s): lane&15 = M row; lane>>4 picks K+0 / K+16.
// Two contiguous 8-element K runs -> two 16B loads, concat via shufflevector.
static __device__ __forceinline__ v16bf load_a_bf16(const __bf16* __restrict__ a,
                                                    int rowbase, int ld, int k0, int lane) {
  const int row = rowbase + (lane & 15);
  const int kb  = (lane >> 4) << 4;           // 0 or 16
  const __bf16* p = a + (size_t)row * ld + k0 + kb;
  v8bf lo = *(const v8bf*)(p);
  v8bf hi = *(const v8bf*)(p + 8);
  return __builtin_shufflevector(lo, hi, 0,1,2,3,4,5,6,7,8,9,10,11,12,13,14,15);
}

// B-fragment (32x16 bf16, contiguous K order k = 16h + s): lane&15 = N col; lane>>4 picks K+0 / K+16.
// wt is the transposed weight, (N,K) row-major: 16 contiguous K -> one aligned 32B load.
static __device__ __forceinline__ v16bf load_b_bf16(const __bf16* __restrict__ wt,
                                                    int colbase, int K, int k0, int lane) {
  const int col = colbase + (lane & 15);
  const int kb  = (lane >> 4) << 4;           // 0 or 16
  return *(const v16bf*)(wt + (size_t)col * K + k0 + kb);
}

static __device__ __forceinline__ float sigmoidf_(float x) {
  return 1.0f / (1.0f + __expf(-x));
}

// ---------- kernel 0a: LDS-tiled transpose + f32->bf16 of a (K,N) weight ----------

__global__ __launch_bounds__(256) void transpose_cvt_kernel(
    const float* __restrict__ src,   // (K, N) row-major f32
    __bf16* __restrict__ dst,        // (N, K) row-major bf16
    int K, int N)
{
  __shared__ __bf16 tile[32][34];    // padded to dodge bank conflicts
  const int tx = threadIdx.x & 31;
  const int ty = threadIdx.x >> 5;   // 0..7
  const int n0 = blockIdx.x * 32;
  const int k0 = blockIdx.y * 32;
#pragma unroll
  for (int i = 0; i < 4; ++i)
    tile[ty + 8 * i][tx] = cvt_bf16(src[(size_t)(k0 + ty + 8 * i) * N + n0 + tx]);
  __syncthreads();
#pragma unroll
  for (int i = 0; i < 4; ++i)
    dst[(size_t)(n0 + ty + 8 * i) * K + k0 + tx] = tile[tx][ty + 8 * i];
}

// ---------- kernel 0b: x -> bf16 (streaming), zero h0 buffers ----------

__global__ void xcvt_zero_kernel(const float* __restrict__ x, __bf16* __restrict__ xbf,
                                 float* __restrict__ hA, __bf16* __restrict__ hbA) {
  const int idx0   = blockIdx.x * blockDim.x + threadIdx.x;
  const int stride = gridDim.x * blockDim.x;
  const int N4 = (B_ * T_ * D_) / 4;
  for (int idx = idx0; idx < N4; idx += stride) {
    f32x4 v = *(const f32x4*)(x + (size_t)idx * 4);
    v4bf o;
#pragma unroll
    for (int i = 0; i < 4; ++i) o[i] = cvt_bf16(v[i]);
    *(v4bf*)(xbf + (size_t)idx * 4) = o;
  }
  const __bf16 z = __builtin_bit_cast(__bf16, (unsigned short)0);
  for (int idx = idx0; idx < B_ * H_; idx += stride) { hA[idx] = 0.0f; hbA[idx] = z; }
}

// ---------- kernel 1: xi = x @ Wi  (32768 x 3072, K=1024), stored as (T,B,3H) ----------

__global__ __launch_bounds__(256) void gru_gemm_xi_kernel(
    const __bf16* __restrict__ xbf,   // (B*T, D) bf16
    const __bf16* __restrict__ WiT,   // (3H, D)  bf16 transposed
    float* __restrict__ xi)           // (T, B, 3H) f32
{
  const int lane = threadIdx.x & 31;
  const int w    = blockIdx.x * 8 + (threadIdx.x >> 5);
  const int NT   = K3H / 64;          // 48 col-tiles of 64
  const int mt   = w / NT;
  const int nt   = w - mt * NT;
  if (mt >= (B_ * T_) / 32) return;
  const int m0 = mt * 32;
  const int n0 = nt * 64;

  v8f c[2][4];
#pragma unroll
  for (int i = 0; i < 2; ++i)
#pragma unroll
    for (int j = 0; j < 4; ++j) c[i][j] = V8F_ZERO;

  for (int k0 = 0; k0 < D_; k0 += 32) {
    v16bf a0 = load_a_bf16(xbf, m0,      D_, k0, lane);
    v16bf a1 = load_a_bf16(xbf, m0 + 16, D_, k0, lane);
    v16bf b0 = load_b_bf16(WiT, n0,      D_, k0, lane);
    v16bf b1 = load_b_bf16(WiT, n0 + 16, D_, k0, lane);
    v16bf b2 = load_b_bf16(WiT, n0 + 32, D_, k0, lane);
    v16bf b3 = load_b_bf16(WiT, n0 + 48, D_, k0, lane);
    c[0][0] = __builtin_amdgcn_wmma_f32_16x16x32_bf16(false, a0, false, b0, (short)0, c[0][0], false, false);
    c[0][1] = __builtin_amdgcn_wmma_f32_16x16x32_bf16(false, a0, false, b1, (short)0, c[0][1], false, false);
    c[0][2] = __builtin_amdgcn_wmma_f32_16x16x32_bf16(false, a0, false, b2, (short)0, c[0][2], false, false);
    c[0][3] = __builtin_amdgcn_wmma_f32_16x16x32_bf16(false, a0, false, b3, (short)0, c[0][3], false, false);
    c[1][0] = __builtin_amdgcn_wmma_f32_16x16x32_bf16(false, a1, false, b0, (short)0, c[1][0], false, false);
    c[1][1] = __builtin_amdgcn_wmma_f32_16x16x32_bf16(false, a1, false, b1, (short)0, c[1][1], false, false);
    c[1][2] = __builtin_amdgcn_wmma_f32_16x16x32_bf16(false, a1, false, b2, (short)0, c[1][2], false, false);
    c[1][3] = __builtin_amdgcn_wmma_f32_16x16x32_bf16(false, a1, false, b3, (short)0, c[1][3], false, false);
  }

  const int cl = lane & 15;
  const int rh = (lane >> 4) * 8;     // C/D row offset per lane-half
#pragma unroll
  for (int i = 0; i < 2; ++i) {
#pragma unroll
    for (int r = 0; r < 8; ++r) {
      const int m = m0 + i * 16 + rh + r;   // row in (B*T)
      const int t = m & (T_ - 1);
      const int b = m >> 9;                 // m / T_
      float* dst = xi + ((size_t)t * B_ + b) * K3H + n0 + cl;
#pragma unroll
      for (int j = 0; j < 4; ++j) dst[j * 16] = c[i][j][r];
    }
  }
}

// ---------- kernel 2: one fused GRU timestep, K split across 2 waves + LDS reduce ----------

__global__ __launch_bounds__(256) void gru_step_kernel(
    const float* __restrict__ xi_t,    // (B, 3H) slab for this t
    const float* __restrict__ h_in,    // f32 (B, H): passthrough term
    const __bf16* __restrict__ hb_in,  // bf16 (B, H): matmul A operand
    const __bf16* __restrict__ WhT,    // (3H, H) bf16 transposed
    const float* __restrict__ bh,      // (3H)
    float* __restrict__ h_out,         // f32 (B, H)
    __bf16* __restrict__ hb_out,       // bf16 (B, H)
    float* __restrict__ out,           // (B, T, H)
    int t)
{
  __shared__ float red[4][3][256];     // [tile-in-block][gate][r*32+lane]
  const int lane = threadIdx.x & 31;
  const int wib  = threadIdx.x >> 5;   // 0..7
  const int p    = wib >> 1;           // tile within block: 0..3
  const int kh   = wib & 1;            // K half
  const int tile = blockIdx.x * 4 + p; // 0..255
  const int mt   = tile >> 6;          // 0..3
  const int nt   = tile & 63;          // 0..63
  const int m0   = mt * 16;
  const int n0   = nt * 16;
  const int kbeg = kh * (H_ / 2);

  v8f cr = V8F_ZERO, cz = V8F_ZERO, cn = V8F_ZERO;

  for (int k0 = kbeg; k0 < kbeg + H_ / 2; k0 += 32) {
    v16bf a  = load_a_bf16(hb_in, m0, H_, k0, lane);
    v16bf br = load_b_bf16(WhT,          n0, H_, k0, lane);
    v16bf bz = load_b_bf16(WhT, H_     + n0, H_, k0, lane);
    v16bf bn = load_b_bf16(WhT, 2 * H_ + n0, H_, k0, lane);
    cr = __builtin_amdgcn_wmma_f32_16x16x32_bf16(false, a, false, br, (short)0, cr, false, false);
    cz = __builtin_amdgcn_wmma_f32_16x16x32_bf16(false, a, false, bz, (short)0, cz, false, false);
    cn = __builtin_amdgcn_wmma_f32_16x16x32_bf16(false, a, false, bn, (short)0, cn, false, false);
  }

  if (kh) {
#pragma unroll
    for (int r = 0; r < 8; ++r) {
      red[p][0][r * 32 + lane] = cr[r];
      red[p][1][r * 32 + lane] = cz[r];
      red[p][2][r * 32 + lane] = cn[r];
    }
  }
  __syncthreads();
  if (kh) return;

#pragma unroll
  for (int r = 0; r < 8; ++r) {
    cr[r] += red[p][0][r * 32 + lane];
    cz[r] += red[p][1][r * 32 + lane];
    cn[r] += red[p][2][r * 32 + lane];
  }

  const int cl = lane & 15;
  const int rh = (lane >> 4) * 8;
  const int j  = n0 + cl;
  const float br_ = bh[j], bz_ = bh[H_ + j], bn_ = bh[2 * H_ + j];
#pragma unroll
  for (int r = 0; r < 8; ++r) {
    const int brow = m0 + rh + r;                 // batch index 0..63
    const size_t xoff = (size_t)brow * K3H + j;
    const float xr = xi_t[xoff];
    const float xz = xi_t[xoff + H_];
    const float xn = xi_t[xoff + 2 * H_];
    const float hp = h_in[(size_t)brow * H_ + j];
    const float rg = sigmoidf_(xr + cr[r] + br_);
    const float zg = sigmoidf_(xz + cz[r] + bz_);
    const float ng = tanhf(xn + rg * (cn[r] + bn_));
    const float hn = (1.0f - zg) * ng + zg * hp;
    h_out[(size_t)brow * H_ + j]  = hn;
    hb_out[(size_t)brow * H_ + j] = cvt_bf16(hn);
    out[(size_t)brow * T_ * H_ + (size_t)t * H_ + j] = hn;
  }
}

// ---------- launch ----------

extern "C" void kernel_launch(void* const* d_in, const int* in_sizes, int n_in,
                              void* d_out, int out_size, void* d_ws, size_t ws_size,
                              hipStream_t stream) {
  const float* x  = (const float*)d_in[0];
  const float* Wi = (const float*)d_in[1];
  const float* Wh = (const float*)d_in[2];
  const float* bh = (const float*)d_in[3];
  float* out = (float*)d_out;

  char* ws = (char*)d_ws;
  float*  xi  = (float*)ws;
  size_t  off = (size_t)T_ * B_ * K3H * sizeof(float);                 // 384 MB
  __bf16* xbf = (__bf16*)(ws + off); off += (size_t)B_ * T_ * D_ * sizeof(__bf16); // 64 MB
  __bf16* WiT = (__bf16*)(ws + off); off += (size_t)K3H * D_ * sizeof(__bf16);
  __bf16* WhT = (__bf16*)(ws + off); off += (size_t)K3H * H_ * sizeof(__bf16);
  float*  hA  = (float*)(ws + off);  off += (size_t)B_ * H_ * sizeof(float);
  float*  hB  = (float*)(ws + off);  off += (size_t)B_ * H_ * sizeof(float);
  __bf16* hbA = (__bf16*)(ws + off); off += (size_t)B_ * H_ * sizeof(__bf16);
  __bf16* hbB = (__bf16*)(ws + off);

  // 0) weights -> bf16 transposed (N,K); x -> bf16; zero h0 (f32 + bf16)
  transpose_cvt_kernel<<<dim3(K3H / 32, D_ / 32), 256, 0, stream>>>(Wi, WiT, D_, K3H);
  transpose_cvt_kernel<<<dim3(K3H / 32, H_ / 32), 256, 0, stream>>>(Wh, WhT, H_, K3H);
  xcvt_zero_kernel<<<2048, 256, 0, stream>>>(x, xbf, hA, hbA);

  // 1) xi = x @ Wi, stored (T,B,3H)
  const int tiles = ((B_ * T_) / 32) * (K3H / 64);                     // 49152 waves
  gru_gemm_xi_kernel<<<tiles / 8, 256, 0, stream>>>(xbf, WiT, xi);

  // 2) 512 fused recurrent steps (ping-pong h buffers, 512 waves/step)
  float*  hin = hA;  float*  hout = hB;
  __bf16* hbin = hbA; __bf16* hbout = hbB;
  for (int t = 0; t < T_; ++t) {
    const float* xit = xi + (size_t)t * B_ * K3H;
    gru_step_kernel<<<64, 256, 0, stream>>>(xit, hin, hbin, WhT, bh, hout, hbout, out, t);
    { float*  tmp = hin;  hin  = hout;  hout  = tmp; }
    { __bf16* tmp = hbin; hbin = hbout; hbout = tmp; }
  }
}
